// Block_377957122290
// MI455X (gfx1250) — matrix operand order repaired and checked by reading, not compiled, not run on hardware
//
#include <hip/hip_runtime.h>
#include <hip/hip_bf16.h>

// ---------------------------------------------------------------------------
// GPT-2 block forward for gfx1250 (MI455X). bf16 WMMA GEMMs, f32 accumulate.
// ---------------------------------------------------------------------------

typedef __attribute__((ext_vector_type(8)))  __bf16 v8bf;
typedef __attribute__((ext_vector_type(16))) __bf16 v16bf;
typedef __attribute__((ext_vector_type(8)))  float  v8f;

__device__ __forceinline__ v8f zero8() {
  v8f z;
#pragma unroll
  for (int i = 0; i < 8; ++i) z[i] = 0.0f;
  return z;
}

// A-fragment (16x32 bf16, MxK): lane half hw holds K {hw*16..hw*16+15}
// -> two b128 loads at p and p+8 elements.
__device__ __forceinline__ v16bf load_a_frag(const __bf16* p) {
  v8bf lo = *(const v8bf*)p;
  v8bf hi = *(const v8bf*)(p + 8);
  v16bf r;
#pragma unroll
  for (int i = 0; i < 8; ++i) { r[i] = lo[i]; r[8 + i] = hi[i]; }
  return r;
}

// B-fragment (32x16 bf16, KxN) from B^T stored [N][K]: lane = column n,
// elements = 16 contiguous K values starting at hw*16 -> one 32B load.
__device__ __forceinline__ v16bf load_b_frag(const __bf16* p) {
  return *(const v16bf*)p;
}

__device__ __forceinline__ v8f wmma_bf16(v16bf a, v16bf b, v8f c) {
  return __builtin_amdgcn_wmma_f32_16x16x32_bf16(
      false, a, false, b, (short)0, c, false, false);
}

__device__ __forceinline__ float gelu_tanh(float u) {
  return 0.5f * u * (1.0f + tanhf(0.7978845608028654f * (u + 0.044715f * u * u * u)));
}

// ---------------------------------------------------------------------------
// LayerNorm (f32 in) -> bf16 out. One block per row of length C (C == 1024).
// The row is staged into LDS with GLOBAL_LOAD_ASYNC_TO_LDS_B128 (ASYNCcnt
// path): 256 lanes x 16 B = 4 KB row. Low 32 bits of a generic pointer to
// __shared__ are the wave-relative LDS offset (ISA 10.2 aperture mapping).
// ---------------------------------------------------------------------------
__global__ __launch_bounds__(256)
void ln_kernel(const float* __restrict__ x, const float* __restrict__ w,
               const float* __restrict__ b, __bf16* __restrict__ out, int C) {
  const int row = blockIdx.x;
  const int tid = threadIdx.x;
  const float* xr = x + (size_t)row * C;

  __shared__ float xs[1024];
  __shared__ float sh0[256];
  __shared__ float sh1[256];

  {
    unsigned lds_off = (unsigned)(unsigned long long)(uintptr_t)(&xs[tid * 4]);
    const float* g = xr + tid * 4;
    asm volatile("global_load_async_to_lds_b128 %0, %1, off"
                 :: "v"(lds_off), "v"(g)
                 : "memory");
    asm volatile("s_wait_asynccnt 0" ::: "memory");
  }
  __syncthreads();

  float s = 0.0f, s2 = 0.0f;
  for (int i = tid; i < C; i += 256) {
    float v = xs[i];
    s += v; s2 += v * v;
  }
  sh0[tid] = s; sh1[tid] = s2;
  __syncthreads();
  for (int off = 128; off > 0; off >>= 1) {
    if (tid < off) { sh0[tid] += sh0[tid + off]; sh1[tid] += sh1[tid + off]; }
    __syncthreads();
  }
  const float mu  = sh0[0] / (float)C;
  const float var = sh1[0] / (float)C - mu * mu;
  const float inv = rsqrtf(var + 1e-5f);
  __bf16* orow = out + (size_t)row * C;
  for (int i = tid; i < C; i += 256)
    orow[i] = (__bf16)((xs[i] - mu) * inv * w[i] + b[i]);
}

// ---------------------------------------------------------------------------
// Transpose + convert: f32 [K][N] row-major -> bf16 [N][K] row-major.
// ---------------------------------------------------------------------------
__global__ void tcvt_kernel(const float* __restrict__ in, __bf16* __restrict__ out,
                            int K, int N) {
  const size_t n = (size_t)K * (size_t)N;
  const size_t stride = (size_t)gridDim.x * blockDim.x;
  for (size_t idx = (size_t)blockIdx.x * blockDim.x + threadIdx.x; idx < n; idx += stride) {
    size_t k = idx / (size_t)N;
    size_t c = idx - k * (size_t)N;
    out[c * (size_t)K + k] = (__bf16)in[idx];
  }
}

// ---------------------------------------------------------------------------
// Extract V from qkv and transpose: vT[bh][d][t] = qkv[(b*T+t)*3C + 2C + h*64 + d]
// ---------------------------------------------------------------------------
__global__ void vtr_kernel(const __bf16* __restrict__ qkv, __bf16* __restrict__ vT) {
  const size_t n = (size_t)32 * 64 * 2048;  // B*H * D * T
  const size_t stride = (size_t)gridDim.x * blockDim.x;
  for (size_t idx = (size_t)blockIdx.x * blockDim.x + threadIdx.x; idx < n; idx += stride) {
    int t  = (int)(idx & 2047);
    int d  = (int)((idx >> 11) & 63);
    int bh = (int)(idx >> 17);
    int b = bh >> 4, head = bh & 15;
    vT[idx] = qkv[((size_t)(b * 2048 + t)) * 3072 + 2048 + head * 64 + d];
  }
}

// ---------------------------------------------------------------------------
// WMMA GEMM: out[M,N] = A[M,K](bf16) * B (given as B^T, bf16 [N][K]) + bias.
// EPI 0: -> bf16 out.  EPI 1: GELU -> bf16 out.  EPI 2: + resid -> f32 out.
// Block (32,4): 4 waves, each wave produces a 32x64 strip (two M-tiles sharing
// each B fragment -> 8 WMMA per 8 b128 loads). Grid (N/256, M/32).
// ---------------------------------------------------------------------------
template <int EPI>
__global__ __launch_bounds__(128)
void gemm_bf16_wmma(const __bf16* __restrict__ A, const __bf16* __restrict__ BT,
                    const float* __restrict__ bias, const float* __restrict__ resid,
                    float* __restrict__ outf, __bf16* __restrict__ outb,
                    int M, int N, int K) {
  const int lane = threadIdx.x;
  const int hw = lane >> 4;
  const int lr = lane & 15;
  const int m0 = blockIdx.y * 32;
  const int nb = blockIdx.x * 256 + threadIdx.y * 64;

  v8f acc[2][4];
#pragma unroll
  for (int mi = 0; mi < 2; ++mi)
#pragma unroll
    for (int j = 0; j < 4; ++j) acc[mi][j] = zero8();

  const __bf16* ap0 = A + (size_t)(m0 + lr) * K + hw * 16;
  const __bf16* ap1 = ap0 + (size_t)16 * K;
  const __bf16* bp  = BT + (size_t)(nb + lr) * K + hw * 16;

  for (int k0 = 0; k0 < K; k0 += 32) {
    // Prefetch a few K-tiles ahead (L2-resident working set; speculative,
    // OOB tail prefetches are silently dropped).
    __builtin_prefetch(ap0 + k0 + 512, 0, 0);
    __builtin_prefetch(ap1 + k0 + 512, 0, 0);
    __builtin_prefetch(bp + k0 + 512, 0, 0);

    v16bf af0 = load_a_frag(ap0 + k0);
    v16bf af1 = load_a_frag(ap1 + k0);
#pragma unroll
    for (int j = 0; j < 4; ++j) {
      v16bf bf_ = load_b_frag(bp + (size_t)(j * 16) * K + k0);
      acc[0][j] = wmma_bf16(af0, bf_, acc[0][j]);
      acc[1][j] = wmma_bf16(af1, bf_, acc[1][j]);
    }
  }

#pragma unroll
  for (int mi = 0; mi < 2; ++mi)
#pragma unroll
    for (int j = 0; j < 4; ++j) {
      const int col = nb + j * 16 + lr;
      const float bv = bias[col];
#pragma unroll
      for (int i = 0; i < 8; ++i) {
        const int row = m0 + mi * 16 + hw * 8 + i;
        float v = acc[mi][j][i] + bv;
        const size_t idx = (size_t)row * N + col;
        if (EPI == 1) v = gelu_tanh(v);
        if (EPI == 2) outf[idx] = v + resid[idx];
        else          outb[idx] = (__bf16)v;
      }
    }
}

// ---------------------------------------------------------------------------
// Flash-style attention (no causal mask, matching reference).
// qkv: [B*T, 3C] bf16.  vT: [B*H][D=64][T] bf16.  y: [B*T, C] bf16.
// Block (32,8): 8 waves, each wave owns 16 query rows. Grid (T/16/8, B*H).
// ---------------------------------------------------------------------------
__global__ __launch_bounds__(256)
void attn_fa_kernel(const __bf16* __restrict__ qkv, const __bf16* __restrict__ vT,
                    __bf16* __restrict__ y) {
  const int T = 2048, C3 = 3072, Cc = 1024, D = 64;
  const int lane = threadIdx.x;
  const int hw = lane >> 4;
  const int lr = lane & 15;
  const int bh = blockIdx.y;
  const int b = bh >> 4, head = bh & 15;
  const int mb = blockIdx.x * 8 + threadIdx.y;
  const int m0 = mb * 16;

  const __bf16* qb = qkv + (size_t)b * T * C3 + (size_t)head * D;
  const __bf16* kb = qb + Cc;
  const __bf16* vb = vT + (size_t)bh * D * T;

  __shared__ __bf16 pbuf[8][16][32];
  __bf16(*P)[32] = pbuf[threadIdx.y];

  // Q A-fragments for the full D=64 (two K=32 chunks), reused every iteration.
  v16bf qf[2];
#pragma unroll
  for (int kk = 0; kk < 2; ++kk)
    qf[kk] = load_a_frag(qb + (size_t)(m0 + lr) * C3 + kk * 32 + hw * 16);

  v8f o[4];
#pragma unroll
  for (int j = 0; j < 4; ++j) o[j] = zero8();
  float rm[8], rl[8];
#pragma unroll
  for (int i = 0; i < 8; ++i) { rm[i] = -1e30f; rl[i] = 0.0f; }

  for (int t0 = 0; t0 < T; t0 += 32) {
    // S tile: 16 rows x 32 key-columns, two 16x16 accumulators.
    v8f s0 = zero8(), s1 = zero8();
#pragma unroll
    for (int kk = 0; kk < 2; ++kk) {
      v16bf kf0 = load_b_frag(kb + (size_t)(t0 + lr) * C3 + kk * 32 + hw * 16);
      s0 = wmma_bf16(qf[kk], kf0, s0);
      v16bf kf1 = load_b_frag(kb + (size_t)(t0 + 16 + lr) * C3 + kk * 32 + hw * 16);
      s1 = wmma_bf16(qf[kk], kf1, s1);
    }

    const float sc = 0.125f;  // 1/sqrt(64)
    float p0[8], p1[8], mx[8];
#pragma unroll
    for (int i = 0; i < 8; ++i) {
      p0[i] = s0[i] * sc;
      p1[i] = s1[i] * sc;
      mx[i] = fmaxf(p0[i], p1[i]);
    }
    // Row max across the 16 lanes of each half-wave (xor masks stay in-half).
#pragma unroll
    for (int i = 0; i < 8; ++i) {
#pragma unroll
      for (int m = 1; m <= 8; m <<= 1)
        mx[i] = fmaxf(mx[i], __shfl_xor(mx[i], m, 32));
    }
    float corr[8], rs[8];
#pragma unroll
    for (int i = 0; i < 8; ++i) {
      float mn = fmaxf(rm[i], mx[i]);
      corr[i] = __expf(rm[i] - mn);
      rm[i] = mn;
      p0[i] = __expf(p0[i] - mn);
      p1[i] = __expf(p1[i] - mn);
      rs[i] = p0[i] + p1[i];
    }
#pragma unroll
    for (int i = 0; i < 8; ++i) {
#pragma unroll
      for (int m = 1; m <= 8; m <<= 1)
        rs[i] += __shfl_xor(rs[i], m, 32);
      rl[i] = rl[i] * corr[i] + rs[i];
    }
#pragma unroll
    for (int j = 0; j < 4; ++j)
#pragma unroll
      for (int i = 0; i < 8; ++i) o[j][i] *= corr[i];

    // Bounce P through LDS to reshuffle C-layout -> A-fragment layout.
#pragma unroll
    for (int i = 0; i < 8; ++i) {
      P[hw * 8 + i][lr]      = (__bf16)p0[i];
      P[hw * 8 + i][16 + lr] = (__bf16)p1[i];
    }
    v16bf pf = load_a_frag(&P[lr][hw * 16]);

    // O += P (16x32) @ V (32x64): 4 d-tiles.
#pragma unroll
    for (int j = 0; j < 4; ++j) {
      v16bf vf = load_b_frag(vb + (size_t)(j * 16 + lr) * T + t0 + hw * 16);
      o[j] = wmma_bf16(pf, vf, o[j]);
    }
  }

#pragma unroll
  for (int j = 0; j < 4; ++j)
#pragma unroll
    for (int i = 0; i < 8; ++i) {
      const int tq = m0 + hw * 8 + i;
      const size_t idx = ((size_t)(b * T + tq)) * Cc + head * D + j * 16 + lr;
      y[idx] = (__bf16)(o[j][i] / rl[i]);
    }
}

// ---------------------------------------------------------------------------
// Launcher
// ---------------------------------------------------------------------------
extern "C" void kernel_launch(void* const* d_in, const int* in_sizes, int n_in,
                              void* d_out, int out_size, void* d_ws, size_t ws_size,
                              hipStream_t stream) {
  (void)in_sizes; (void)n_in; (void)out_size; (void)ws_size;

  const int T = 2048, C = 1024;
  const int M = 2 * T;        // 4096 rows
  const int C3 = 3 * C;       // 3072
  const int C4 = 4 * C;       // 4096

  const float* x       = (const float*)d_in[0];
  const float* ln1_w   = (const float*)d_in[1];
  const float* ln1_b   = (const float*)d_in[2];
  const float* attn_w  = (const float*)d_in[3];
  const float* attn_b  = (const float*)d_in[4];
  const float* proj_w  = (const float*)d_in[5];
  const float* proj_b  = (const float*)d_in[6];
  const float* ln2_w   = (const float*)d_in[7];
  const float* ln2_b   = (const float*)d_in[8];
  const float* fc_w    = (const float*)d_in[9];
  const float* fc_b    = (const float*)d_in[10];
  const float* mlp_w   = (const float*)d_in[11];
  const float* mlp_b   = (const float*)d_in[12];
  float* out = (float*)d_out;

  // Workspace layout (bytes, 256-aligned regions).
  char* ws = (char*)d_ws;
  size_t off = 0;
  auto alloc = [&](size_t bytes) -> void* {
    void* p = ws + off;
    off += (bytes + 255) & ~(size_t)255;
    return p;
  };
  __bf16* wT_attn = (__bf16*)alloc((size_t)C3 * C * 2);   //  6 MB  [3C][C]
  __bf16* wT_proj = (__bf16*)alloc((size_t)C * C * 2);    //  2 MB  [C][C]
  __bf16* wT_fc   = (__bf16*)alloc((size_t)C4 * C * 2);   //  8 MB  [4C][C]
  __bf16* wT_mlp  = (__bf16*)alloc((size_t)C * C4 * 2);   //  8 MB  [C][4C]
  __bf16* hbuf    = (__bf16*)alloc((size_t)M * C * 2);    //  8 MB  ln1/ln2 out
  float*  x1      = (float*)alloc((size_t)M * C * 4);     // 16 MB  attn residual out
  __bf16* qkv     = (__bf16*)alloc((size_t)M * C3 * 2);   // 24 MB
  __bf16* vTb     = (__bf16*)alloc((size_t)M * C * 2);    //  8 MB  [B*H][64][T]
  __bf16* ybuf    = (__bf16*)alloc((size_t)M * C * 2);    //  8 MB
  // hfc (32 MB) aliases the dead qkv+vT+y region (40 MB) after attention.
  __bf16* hfc = qkv;

  const dim3 gblk(32, 4);

  // 1) Weight transposes + bf16 convert.
  tcvt_kernel<<<dim3(8192), dim3(256), 0, stream>>>(attn_w, wT_attn, C, C3);
  tcvt_kernel<<<dim3(4096), dim3(256), 0, stream>>>(proj_w, wT_proj, C, C);
  tcvt_kernel<<<dim3(8192), dim3(256), 0, stream>>>(fc_w,   wT_fc,   C, C4);
  tcvt_kernel<<<dim3(8192), dim3(256), 0, stream>>>(mlp_w,  wT_mlp,  C4, C);

  // 2) LN1 -> bf16 (async-to-LDS staging).
  ln_kernel<<<dim3(M), dim3(256), 0, stream>>>(x, ln1_w, ln1_b, hbuf, C);

  // 3) QKV GEMM: [M,C]x[C,3C] + bias -> bf16 qkv.
  gemm_bf16_wmma<0><<<dim3(C3 / 256, M / 32), gblk, 0, stream>>>(
      hbuf, wT_attn, attn_b, nullptr, nullptr, qkv, M, C3, C);

  // 4) V transpose for PV WMMA.
  vtr_kernel<<<dim3(8192), dim3(256), 0, stream>>>(qkv, vTb);

  // 5) Attention -> y (bf16).
  attn_fa_kernel<<<dim3(T / 16 / 8, 32), dim3(32, 8), 0, stream>>>(qkv, vTb, ybuf);

  // 6) Attn proj + residual -> x1 (f32).
  gemm_bf16_wmma<2><<<dim3(C / 256, M / 32), gblk, 0, stream>>>(
      ybuf, wT_proj, proj_b, x, x1, nullptr, M, C, C);

  // 7) LN2 -> bf16.
  ln_kernel<<<dim3(M), dim3(256), 0, stream>>>(x1, ln2_w, ln2_b, hbuf, C);

  // 8) FC GEMM + GELU -> bf16 hfc.
  gemm_bf16_wmma<1><<<dim3(C4 / 256, M / 32), gblk, 0, stream>>>(
      hbuf, wT_fc, fc_b, nullptr, nullptr, hfc, M, C4, C);

  // 9) MLP proj + residual -> f32 out.
  gemm_bf16_wmma<2><<<dim3(C / 256, M / 32), gblk, 0, stream>>>(
      hfc, wT_mlp, mlp_b, x1, out, nullptr, M, C, C4);
}
